// LAS_10058813407380
// MI455X (gfx1250) — compile-verified, matrix operand order checked by reading, not run on hardware
//
#include <hip/hip_runtime.h>

typedef __bf16 bf16;
typedef __attribute__((ext_vector_type(16))) __bf16 bf16x16;
typedef __attribute__((ext_vector_type(8)))  float  floatx8;

#define TT 256          // encoder time steps after conv (T')
#define NB 16           // batch
#define HSZ 1024        // LSTM hidden
#define DECT 100        // decoder steps
#define FEAT 2048       // 2*ENC_H

static inline unsigned cdiv(size_t a, size_t b) { return (unsigned)((a + b - 1) / b); }

// ---------------------------------------------------------------------------
// WMMA helper: accumulate a 16x16 f32 tile over K (bf16 A [16,lda], B = W rows)
// A lane (l) mapping per CDNA5 ISA: row = l&15, K-chunks at kbase=(l<16?0:8)
// ---------------------------------------------------------------------------
__device__ __forceinline__ floatx8 wmma_tile(floatx8 acc, const bf16* A, int lda,
                                             const bf16* Bw, int ldb, int K) {
  int lane = threadIdx.x & 31;
  int rc = lane & 15;
  int kb = (lane < 16) ? 0 : 8;
  const bf16* ap = A + (size_t)rc * lda + kb;
  const bf16* bp = Bw + (size_t)rc * ldb + kb;
  for (int k0 = 0; k0 < K; k0 += 32) {
    union { bf16x16 v; uint4 u[2]; } a, b;
    a.u[0] = *reinterpret_cast<const uint4*>(ap + k0);
    a.u[1] = *reinterpret_cast<const uint4*>(ap + k0 + 16);
    b.u[0] = *reinterpret_cast<const uint4*>(bp + k0);
    b.u[1] = *reinterpret_cast<const uint4*>(bp + k0 + 16);
    acc = __builtin_amdgcn_wmma_f32_16x16x32_bf16(false, a.v, false, b.v,
                                                  (short)0, acc, false, false);
  }
  return acc;
}

__device__ __forceinline__ float sigmf(float x) { return 1.f / (1.f + __expf(-x)); }

// ---------------------------------------------------------------------------
// Fused LSTM step: gates = xt@Wih^T + h@Whh^T + b ; cell update; h out (bf16)
// grid.x = 64*ndir blocks; block = 256 threads = 8 waves:
//   wave = gate(0..3) + 4*khalf ; each wave accumulates half the K reduction
//   for its gate's 16x16 tile; halves are summed through LDS in the fused
//   activation phase (doubles in-flight waves streaming weights from L2).
// ---------------------------------------------------------------------------
struct LstmDir {
  const bf16* xt;  int Dx;
  const bf16* hprev;
  const bf16* Wih; const bf16* Whh; const float* bias;
  float* c; bf16* hnext;
  bf16* seqout; int seq_ld;
};

__global__ __launch_bounds__(256) void lstm_step_kernel(LstmDir p0, LstmDir p1) {
  __shared__ float g[8][16][16];
  const LstmDir p = (blockIdx.x >= 64) ? p1 : p0;
  int h0 = (blockIdx.x & 63) * 16;
  int wave = threadIdx.x >> 5;
  int gate = wave & 3;
  int kh = wave >> 2;                        // K-split half: 0 or 1
  int lane = threadIdx.x & 31;
  int rc = lane & 15;
  int col0 = gate * HSZ + h0;                // gate column tile (i/f/g/o)
  float bv = (kh == 0) ? p.bias[col0 + rc] : 0.f;
  floatx8 acc;
#pragma unroll
  for (int i = 0; i < 8; ++i) acc[i] = bv;
  int dxh = p.Dx >> 1;                       // all Dx are multiples of 64
  acc = wmma_tile(acc, p.xt + kh * dxh, p.Dx,
                  p.Wih + (size_t)col0 * p.Dx + kh * dxh, p.Dx, dxh);
  acc = wmma_tile(acc, p.hprev + kh * (HSZ / 2), HSZ,
                  p.Whh + (size_t)col0 * HSZ + kh * (HSZ / 2), HSZ, HSZ / 2);
  int moff = (lane < 16) ? 0 : 8;
#pragma unroll
  for (int v = 0; v < 8; ++v) g[wave][v + moff][rc] = acc[v];
  __syncthreads();
  {
    int e = threadIdx.x;                     // 256 threads, 256 elements
    int m = e >> 4, n = e & 15;
    float gi = g[0][m][n] + g[4][m][n];
    float gf = g[1][m][n] + g[5][m][n];
    float gg = g[2][m][n] + g[6][m][n];
    float go = g[3][m][n] + g[7][m][n];
    int ci = m * HSZ + h0 + n;
    float cn = sigmf(gf) * p.c[ci] + sigmf(gi) * tanhf(gg);
    float hn = sigmf(go) * tanhf(cn);
    p.c[ci] = cn;
    p.hnext[ci] = (bf16)hn;
    if (p.seqout) p.seqout[m * p.seq_ld + h0 + n] = (bf16)hn;
  }
}

// ---------------------------------------------------------------------------
// h_proj = encode_feature @ att_Wh^T + att_bh ; feat is time-major [T][16][2048]
// grid(256 row-tiles, 32 col-groups), 4 waves per block
// ---------------------------------------------------------------------------
__global__ __launch_bounds__(128) void hproj_kernel(const bf16* feat, const bf16* Wh,
                                                    const float* bh, bf16* hproj) {
  int b = blockIdx.x >> 4;
  int t0 = (blockIdx.x & 15) * 16;
  int wave = threadIdx.x >> 5, lane = threadIdx.x & 31;
  int n0 = (blockIdx.y * 4 + wave) * 16;
  int rc = lane & 15, moff = (lane < 16) ? 0 : 8;
  float bv = bh[n0 + rc];
  floatx8 acc;
#pragma unroll
  for (int i = 0; i < 8; ++i) acc[i] = bv;
  const bf16* A = feat + ((size_t)t0 * NB + b) * FEAT;   // row stride = 16*2048
  acc = wmma_tile(acc, A, NB * FEAT, Wh + (size_t)(n0)*FEAT, FEAT, FEAT);
#pragma unroll
  for (int v = 0; v < 8; ++v)
    hproj[((size_t)b * TT + t0 + v + moff) * FEAT + n0 + rc] = (bf16)acc[v];
}

// out[16,N](bf16) = A[16,K](bf16) @ W[N,K]^T + bias ; grid N/64, block 128
__global__ __launch_bounds__(128) void linear16_kernel(const bf16* A, int lda,
                                                       const bf16* W, const float* bias,
                                                       bf16* out, int N, int K) {
  int wave = threadIdx.x >> 5, lane = threadIdx.x & 31;
  int n0 = (blockIdx.x * 4 + wave) * 16;
  int rc = lane & 15, moff = (lane < 16) ? 0 : 8;
  float bv = bias[n0 + rc];
  floatx8 acc;
#pragma unroll
  for (int i = 0; i < 8; ++i) acc[i] = bv;
  acc = wmma_tile(acc, A, lda, W + (size_t)n0 * K, K, K);
#pragma unroll
  for (int v = 0; v < 8; ++v) out[(size_t)(v + moff) * N + n0 + rc] = (bf16)acc[v];
}

// scores[b,t] = sum_d hproj[b,t,d] * sproj[b,d] ; one wave per 8 t values
// vectorized: each lane reads 8 bf16 (uint4) per step, wave covers 256 dims
__global__ void scores_kernel(const bf16* hproj, const bf16* sproj, float* scores) {
  int b = blockIdx.x, wave = threadIdx.x >> 5, lane = threadIdx.x & 31;
  const bf16* sp = sproj + (size_t)b * FEAT;
  for (int i = 0; i < 8; ++i) {
    int t = blockIdx.y * 32 + wave * 8 + i;
    const bf16* hp = hproj + ((size_t)b * TT + t) * FEAT;
    float s = 0.f;
#pragma unroll
    for (int j = 0; j < 8; ++j) {
      int d = j * 256 + lane * 8;
      union { uint4 u; bf16 h[8]; } xa, xb;
      xa.u = *reinterpret_cast<const uint4*>(hp + d);
      xb.u = *reinterpret_cast<const uint4*>(sp + d);
#pragma unroll
      for (int e = 0; e < 8; ++e) s += (float)xa.h[e] * (float)xb.h[e];
    }
    for (int off = 16; off; off >>= 1) s += __shfl_xor(s, off, 32);
    if (lane == 0) scores[b * TT + t] = s;
  }
}

__global__ void softmax_kernel(const float* scores, float* attw) {
  __shared__ float red[TT];
  int b = blockIdx.x, t = threadIdx.x;
  float v = scores[b * TT + t];
  red[t] = v; __syncthreads();
  for (int s = 128; s; s >>= 1) { if (t < s) red[t] = fmaxf(red[t], red[t + s]); __syncthreads(); }
  float mx = red[0]; __syncthreads();
  float e = __expf(v - mx);
  red[t] = e; __syncthreads();
  for (int s = 128; s; s >>= 1) { if (t < s) red[t] += red[t + s]; __syncthreads(); }
  attw[b * TT + t] = e / red[0];
}

// ctx[b,d] = sum_t attw[b,t]*hproj[b,t,d]; writes f32 contexts + bf16 dec input
__global__ void ctx_kernel(const bf16* hproj, const float* attw, float* ctx_out,
                           bf16* dinp, int step) {
  int b = blockIdx.x;
  int d = blockIdx.y * 128 + threadIdx.x;
  const bf16* hp = hproj + (size_t)b * TT * FEAT + d;
  const float* aw = attw + b * TT;
  float acc = 0.f;
  for (int t = 0; t < TT; ++t) acc += aw[t] * (float)hp[(size_t)t * FEAT];
  ctx_out[((size_t)b * DECT + step) * FEAT + d] = acc;
  dinp[b * 2560 + 512 + d] = (bf16)acc;
}

__global__ void embseq_kernel(const int* text, const float* embt, bf16* embs) {
  int s = blockIdx.x, b = blockIdx.y, e = threadIdx.x;
  int tok = (s == 0) ? 0 : text[b * DECT + (s - 1)];
  embs[((size_t)s * NB + b) * 512 + e] = (bf16)embt[(size_t)tok * 512 + e];
}

__global__ void inp_emb_kernel(const bf16* embs, bf16* dinp, int s) {
  int b = blockIdx.x, e = threadIdx.x;
  dinp[b * 2560 + e] = embs[((size_t)s * NB + b) * 512 + e];
}

// char head + log_softmax: block (32 vocab lanes, 16 batch waves)
__global__ void char_kernel(const bf16* h, const float* W, const float* cb,
                            float* logp, int step) {
  int v = threadIdx.x, b = threadIdx.y;
  const bf16* hr = h + (size_t)b * HSZ;
  const float* wr = W + (size_t)v * HSZ;
  float s = cb[v];
  for (int k = 0; k < HSZ; ++k) s += (float)hr[k] * wr[k];
  float mx = s;
  for (int off = 16; off; off >>= 1) mx = fmaxf(mx, __shfl_xor(mx, off, 32));
  float e = __expf(s - mx);
  float sum = e;
  for (int off = 16; off; off >>= 1) sum += __shfl_xor(sum, off, 32);
  logp[((size_t)b * DECT + step) * 32 + v] = s - mx - __logf(sum);
}

// ---------------------------------------------------------------------------
// Conv front-end
// ---------------------------------------------------------------------------
__global__ void conv1_kernel(const float* spec, const float* w, const float* bias, float* y) {
  int idx = blockIdx.x * 256 + threadIdx.x;
  if (idx >= 16 * 32 * 512 * 20) return;
  int fo = idx % 20, to = (idx / 20) % 512, oc = (idx / (20 * 512)) % 32, b = idx / (20 * 512 * 32);
  float acc = bias[oc];
  for (int ic = 0; ic < 2; ++ic)
    for (int kh = 0; kh < 3; ++kh) {
      int t = 2 * to - 1 + kh;
      if ((unsigned)t >= 1024u) continue;
      for (int kw = 0; kw < 3; ++kw) {
        int f = 2 * fo - 1 + kw;
        if ((unsigned)f >= 40u) continue;
        acc += w[((oc * 2 + ic) * 3 + kh) * 3 + kw] *
               spec[((size_t)b * 80 + ic * 40 + f) * 1024 + t];
      }
    }
  y[idx] = acc;
}

__global__ void conv2_kernel(const float* x, const float* w, const float* bias, float* y) {
  int idx = blockIdx.x * 256 + threadIdx.x;
  if (idx >= 16 * 32 * 256 * 10) return;
  int fo = idx % 10, to = (idx / 10) % 256, oc = (idx / 2560) % 32, b = idx / 81920;
  float acc = bias[oc];
  for (int ic = 0; ic < 32; ++ic)
    for (int kh = 0; kh < 3; ++kh) {
      int t = 2 * to - 1 + kh;
      if ((unsigned)t >= 512u) continue;
      for (int kw = 0; kw < 3; ++kw) {
        int f = 2 * fo - 1 + kw;
        if ((unsigned)f >= 20u) continue;
        acc += w[((oc * 32 + ic) * 3 + kh) * 3 + kw] *
               x[(((size_t)b * 32 + ic) * 512 + t) * 20 + f];
      }
    }
  y[idx] = acc;
}

__global__ void bnstats_kernel(const float* y, int HW, int C, float* mv) {
  __shared__ float rs[256], rs2[256];
  int c = blockIdx.x;
  float s = 0.f, s2 = 0.f;
  for (int i = threadIdx.x; i < NB * HW; i += 256) {
    int n = i / HW, hw = i % HW;
    float v = y[((size_t)n * C + c) * HW + hw];
    s += v; s2 += v * v;
  }
  rs[threadIdx.x] = s; rs2[threadIdx.x] = s2; __syncthreads();
  for (int st = 128; st; st >>= 1) {
    if (threadIdx.x < st) { rs[threadIdx.x] += rs[threadIdx.x + st]; rs2[threadIdx.x] += rs2[threadIdx.x + st]; }
    __syncthreads();
  }
  if (threadIdx.x == 0) {
    float inv = 1.f / (float)(NB * HW);
    float mean = rs[0] * inv;
    mv[c * 2] = mean;
    mv[c * 2 + 1] = rs2[0] * inv - mean * mean;
  }
}

__global__ void bnrelu_kernel(float* y, const float* mv, const float* g, const float* bb,
                              int C, int HW, int total) {
  int idx = blockIdx.x * 256 + threadIdx.x;
  if (idx >= total) return;
  int c = (idx / HW) % C;
  float v = (y[idx] - mv[c * 2]) * rsqrtf(mv[c * 2 + 1] + 1e-5f) * g[c] + bb[c];
  y[idx] = fmaxf(v, 0.f);
}

// BN2 + ReLU + pack into time-major bf16 x0[t][b][c*10+f]
__global__ void bnrelu_pack_kernel(const float* y, const float* mv, const float* g,
                                   const float* bb, bf16* x0) {
  int idx = blockIdx.x * 256 + threadIdx.x;
  if (idx >= 16 * 32 * 256 * 10) return;
  int f = idx % 10, t = (idx / 10) % 256, c = (idx / 2560) % 32, b = idx / 81920;
  float v = (y[idx] - mv[c * 2]) * rsqrtf(mv[c * 2 + 1] + 1e-5f) * g[c] + bb[c];
  x0[((size_t)t * NB + b) * 320 + c * 10 + f] = (bf16)fmaxf(v, 0.f);
}

__global__ void cvt_kernel(const float* s, bf16* d, int n) {
  int i = blockIdx.x * 256 + threadIdx.x;
  if (i < n) d[i] = (bf16)s[i];
}

__global__ void tail_kernel(const int* text, float* out_text, float* out_len) {
  for (int i = threadIdx.x; i < NB * DECT; i += 256) out_text[i] = (float)text[i];
  if (threadIdx.x < NB) {
    int b = threadIdx.x;
    int first = -1;
    for (int s = 0; s < DECT; ++s) if (text[b * DECT + s] == 0) { first = s; break; }
    out_len[b] = (float)((first <= 0) ? DECT : first);
  }
}

// ---------------------------------------------------------------------------
extern "C" void kernel_launch(void* const* d_in, const int* in_sizes, int n_in,
                              void* d_out, int out_size, void* d_ws, size_t ws_size,
                              hipStream_t stream) {
  (void)in_sizes; (void)n_in; (void)out_size; (void)ws_size;
  const float* spec = (const float*)d_in[0];
  const int*   text = (const int*)d_in[1];
  const float* c1w = (const float*)d_in[4];  const float* c1b = (const float*)d_in[5];
  const float* g1  = (const float*)d_in[6];  const float* b1  = (const float*)d_in[7];
  const float* c2w = (const float*)d_in[8];  const float* c2b = (const float*)d_in[9];
  const float* g2  = (const float*)d_in[10]; const float* b2  = (const float*)d_in[11];
  const float* att_bh = (const float*)d_in[25];
  const float* att_bs = (const float*)d_in[27];
  const float* d0b = (const float*)d_in[30];
  const float* d1b = (const float*)d_in[33];
  const float* chW = (const float*)d_in[34];
  const float* chB = (const float*)d_in[35];
  const float* embt = (const float*)d_in[36];

  char* wp = (char*)d_ws;
  auto alloc = [&](size_t bytes) -> char* {
    char* r = wp; wp += (bytes + 255) & ~(size_t)255; return r;
  };
  auto cvt = [&](int idx, size_t n) -> bf16* {
    bf16* dst = (bf16*)alloc(n * 2);
    cvt_kernel<<<cdiv(n, 256), 256, 0, stream>>>((const float*)d_in[idx], dst, (int)n);
    return dst;
  };

  // bf16 weight conversions (streamed once, stay L2-resident)
  bf16* w_e0f_ih = cvt(12, (size_t)4096 * 320);
  bf16* w_e0f_hh = cvt(13, (size_t)4096 * 1024);
  bf16* w_e0b_ih = cvt(15, (size_t)4096 * 320);
  bf16* w_e0b_hh = cvt(16, (size_t)4096 * 1024);
  bf16* w_e1f_ih = cvt(18, (size_t)4096 * 2048);
  bf16* w_e1f_hh = cvt(19, (size_t)4096 * 1024);
  bf16* w_e1b_ih = cvt(21, (size_t)4096 * 2048);
  bf16* w_e1b_hh = cvt(22, (size_t)4096 * 1024);
  bf16* w_attWh  = cvt(24, (size_t)2048 * 2048);
  bf16* w_attWs  = cvt(26, (size_t)2048 * 1024);
  bf16* w_d0_ih  = cvt(28, (size_t)4096 * 2560);
  bf16* w_d0_hh  = cvt(29, (size_t)4096 * 1024);
  bf16* w_d1_ih  = cvt(31, (size_t)4096 * 1024);
  bf16* w_d1_hh  = cvt(32, (size_t)4096 * 1024);

  // conv front-end
  float* y1 = (float*)alloc((size_t)5242880 * 4);
  float* y2 = (float*)alloc((size_t)1310720 * 4);
  float* mv = (float*)alloc(64 * 4);
  bf16*  x0 = (bf16*)alloc((size_t)1310720 * 2);
  conv1_kernel<<<cdiv(5242880, 256), 256, 0, stream>>>(spec, c1w, c1b, y1);
  bnstats_kernel<<<32, 256, 0, stream>>>(y1, 512 * 20, 32, mv);
  bnrelu_kernel<<<cdiv(5242880, 256), 256, 0, stream>>>(y1, mv, g1, b1, 32, 512 * 20, 5242880);
  conv2_kernel<<<cdiv(1310720, 256), 256, 0, stream>>>(y1, c2w, c2b, y2);
  bnstats_kernel<<<32, 256, 0, stream>>>(y2, 256 * 10, 32, mv);
  bnrelu_pack_kernel<<<cdiv(1310720, 256), 256, 0, stream>>>(y2, mv, g2, b2, x0);

  // encoder buffers
  bf16* hcat0 = (bf16*)alloc((size_t)TT * NB * FEAT * 2);
  bf16* hcat1 = (bf16*)alloc((size_t)TT * NB * FEAT * 2);
  bf16* hst[4];
  for (int i = 0; i < 4; ++i) hst[i] = (bf16*)alloc((size_t)NB * HSZ * 2);
  float* cf = (float*)alloc((size_t)NB * HSZ * 4);
  float* cb = (float*)alloc((size_t)NB * HSZ * 4);

  struct Lay {
    const bf16* xbuf; int Dx;
    const bf16 *wif, *whf, *wib, *whb;
    const float *bf_, *bb_;
    bf16* outb;
  };
  Lay L[2] = {
    { x0,    320,  w_e0f_ih, w_e0f_hh, w_e0b_ih, w_e0b_hh,
      (const float*)d_in[14], (const float*)d_in[17], hcat0 },
    { hcat0, 2048, w_e1f_ih, w_e1f_hh, w_e1b_ih, w_e1b_hh,
      (const float*)d_in[20], (const float*)d_in[23], hcat1 },
  };
  for (int l = 0; l < 2; ++l) {
    hipMemsetAsync(hst[0], 0, (size_t)NB * HSZ * 2, stream);
    hipMemsetAsync(hst[2], 0, (size_t)NB * HSZ * 2, stream);
    hipMemsetAsync(cf, 0, (size_t)NB * HSZ * 4, stream);
    hipMemsetAsync(cb, 0, (size_t)NB * HSZ * 4, stream);
    for (int t = 0; t < TT; ++t) {
      LstmDir f, b;
      f.xt = L[l].xbuf + (size_t)t * NB * L[l].Dx; f.Dx = L[l].Dx;
      f.hprev = hst[t & 1]; f.hnext = hst[(t + 1) & 1];
      f.Wih = L[l].wif; f.Whh = L[l].whf; f.bias = L[l].bf_;
      f.c = cf; f.seqout = L[l].outb + (size_t)t * NB * FEAT; f.seq_ld = FEAT;
      int tb = TT - 1 - t;
      b.xt = L[l].xbuf + (size_t)tb * NB * L[l].Dx; b.Dx = L[l].Dx;
      b.hprev = hst[2 + (t & 1)]; b.hnext = hst[2 + ((t + 1) & 1)];
      b.Wih = L[l].wib; b.Whh = L[l].whb; b.bias = L[l].bb_;
      b.c = cb; b.seqout = L[l].outb + (size_t)tb * NB * FEAT + HSZ; b.seq_ld = FEAT;
      lstm_step_kernel<<<128, 256, 0, stream>>>(f, b);
    }
  }

  // attention projection (hoisted, step-invariant)
  bf16* hproj = (bf16*)alloc((size_t)NB * TT * FEAT * 2);
  hproj_kernel<<<dim3(256, 32), 128, 0, stream>>>(hcat1, w_attWh, att_bh, hproj);

  // embedding sequence (teacher forcing, <sos>=token 0 at step 0)
  bf16* embs = (bf16*)alloc((size_t)DECT * NB * 512 * 2);
  embseq_kernel<<<dim3(DECT, NB), 512, 0, stream>>>(text, embt, embs);

  // decoder buffers
  bf16* sproj = (bf16*)alloc((size_t)NB * FEAT * 2);
  float* scores = (float*)alloc((size_t)NB * TT * 4);
  float* attw   = (float*)alloc((size_t)NB * TT * 4);
  bf16*  dinp   = (bf16*)alloc((size_t)NB * 2560 * 2);
  bf16* h0b[2], *h1b[2];
  for (int i = 0; i < 2; ++i) { h0b[i] = (bf16*)alloc((size_t)NB * HSZ * 2);
                                h1b[i] = (bf16*)alloc((size_t)NB * HSZ * 2); }
  float* c0 = (float*)alloc((size_t)NB * HSZ * 4);
  float* c1 = (float*)alloc((size_t)NB * HSZ * 4);
  hipMemsetAsync(h0b[0], 0, (size_t)NB * HSZ * 2, stream);
  hipMemsetAsync(h1b[0], 0, (size_t)NB * HSZ * 2, stream);
  hipMemsetAsync(c0, 0, (size_t)NB * HSZ * 4, stream);
  hipMemsetAsync(c1, 0, (size_t)NB * HSZ * 4, stream);

  float* out = (float*)d_out;
  float* out_lp = out;                                 // [16,100,32]
  float* out_cx = out + (size_t)NB * DECT * 32;        // [16,100,2048]
  float* out_tx = out_cx + (size_t)NB * DECT * FEAT;   // [16,100]
  float* out_ln = out_tx + (size_t)NB * DECT;          // [16]

  for (int s = 0; s < DECT; ++s) {
    bf16* h0c = h0b[s & 1]; bf16* h0n = h0b[(s + 1) & 1];
    bf16* h1c = h1b[s & 1]; bf16* h1n = h1b[(s + 1) & 1];
    linear16_kernel<<<32, 128, 0, stream>>>(h0c, HSZ, w_attWs, att_bs, sproj, FEAT, HSZ);
    scores_kernel<<<dim3(16, 8), 128, 0, stream>>>(hproj, sproj, scores);
    softmax_kernel<<<16, 256, 0, stream>>>(scores, attw);
    ctx_kernel<<<dim3(16, 16), 128, 0, stream>>>(hproj, attw, out_cx, dinp, s);
    inp_emb_kernel<<<16, 512, 0, stream>>>(embs, dinp, s);
    LstmDir d0;
    d0.xt = dinp; d0.Dx = 2560; d0.hprev = h0c;
    d0.Wih = w_d0_ih; d0.Whh = w_d0_hh; d0.bias = d0b;
    d0.c = c0; d0.hnext = h0n; d0.seqout = nullptr; d0.seq_ld = 0;
    lstm_step_kernel<<<64, 256, 0, stream>>>(d0, d0);
    LstmDir d1;
    d1.xt = h0n; d1.Dx = HSZ; d1.hprev = h1c;
    d1.Wih = w_d1_ih; d1.Whh = w_d1_hh; d1.bias = d1b;
    d1.c = c1; d1.hnext = h1n; d1.seqout = nullptr; d1.seq_ld = 0;
    lstm_step_kernel<<<64, 256, 0, stream>>>(d1, d1);
    char_kernel<<<1, dim3(32, 16), 0, stream>>>(h1n, chW, chB, out_lp, s);
  }

  tail_kernel<<<1, 256, 0, stream>>>(text, out_tx, out_ln);
}
